// Net_89343909691628
// MI455X (gfx1250) — matrix operand order repaired and checked by reading, not compiled, not run on hardware
//
#include <hip/hip_runtime.h>

typedef __attribute__((ext_vector_type(16))) __bf16 v16bf;
typedef __attribute__((ext_vector_type(8)))  float  v8f;
typedef __attribute__((ext_vector_type(4)))  float  v4f;

#define BATCH   32768
#define XCOLS   3072      // flattened 3*32*32 row length of x
#define KIN     1024      // capped input dim
#define EMBED   16
#define TINY    16
#define NEXP    4
#define OUTD    1000
#define KP      96        // padded K for 2nd GEMM: 64 hg + 4 gate + 28 zero (3 x K=32)
#define NTILES  63        // ceil(1000/16) -> padded N = 1008

union V16U { v16bf v; __bf16 e[16]; uint4 q[2]; };

__device__ __forceinline__ v8f wmma_bf16(v16bf a, v16bf b, v8f c) {
  // v_wmma_f32_16x16x32_bf16 (neg_a, A, neg_b, B, c_mod, C, reuse_a, reuse_b)
  return __builtin_amdgcn_wmma_f32_16x16x32_bf16(false, a, false, b, (short)0, c,
                                                 false, false);
}

__device__ __forceinline__ void cvt8(__bf16* d, v4f a, v4f b) {
  d[0]=(__bf16)a[0]; d[1]=(__bf16)a[1]; d[2]=(__bf16)a[2]; d[3]=(__bf16)a[3];
  d[4]=(__bf16)b[0]; d[5]=(__bf16)b[1]; d[6]=(__bf16)b[2]; d[7]=(__bf16)b[3];
}

__device__ __forceinline__ v4f ldnt4(const float* p) {
  return __builtin_nontemporal_load((const v4f*)p);   // streaming x: TH=NT
}

__device__ __forceinline__ v4f ld4(const float* p) {
  return *(const v4f*)p;
}

// --------------------------------------------------------------------------
// k0: pack W2 [4][16][1000] + b2 [4][1000] into bf16 W2augT [1008][96]:
//   W2augT[n][k] = W2[k/16][k%16][n]  (k<64)
//                = b2[k-64][n]        (64<=k<68)
//                = 0                  (k>=68 or n>=1000)
// --------------------------------------------------------------------------
__global__ __launch_bounds__(256) void k0_pack_w2(
    const float* __restrict__ W2, const float* __restrict__ b2,
    unsigned short* __restrict__ w2t_raw)
{
  __bf16* w2t = reinterpret_cast<__bf16*>(w2t_raw);
  int idx = blockIdx.x * 256 + threadIdx.x;
  if (idx >= NTILES * 16 * KP) return;
  int n = idx / KP;
  int k = idx - n * KP;
  float v = 0.f;
  if (n < OUTD) {
    if (k < 64)       v = W2[(size_t)k * OUTD + n];          // (e*16+h)*1000 + n
    else if (k < 68)  v = b2[(size_t)(k - 64) * OUTD + n];
  }
  w2t[idx] = (__bf16)v;
}

// --------------------------------------------------------------------------
// k1: per wave, one 16-row tile.
//   e = relu(x[:, :1024] @ We^T + be)  via 32x v_wmma_f32_16x16x32_bf16
//   gate = softmax(e @ Wg^T + bg); h = relu(e @ W1 + b1) per expert (VALU)
//   hg[b, 0:64] = gate_e * h_e ; hg[b, 64:68] = gate ; hg[b, 68:96] = 0 (bf16)
// K-loop manually unrolled 2x with all 16 loads issued before any cvt/WMMA
// to deepen outstanding-load count (HBM latency hiding with only 2048 waves).
// --------------------------------------------------------------------------
__global__ __launch_bounds__(256) void k1_embed_gate_hg(
    const float* __restrict__ x,  const float* __restrict__ We,
    const float* __restrict__ be, const float* __restrict__ W1,
    const float* __restrict__ b1, const float* __restrict__ Wg,
    const float* __restrict__ bg, unsigned short* __restrict__ hg_raw)
{
  __bf16* hg = reinterpret_cast<__bf16*>(hg_raw);
  __shared__ float eLds[8][16][16];
  __shared__ float gLds[8][16][4];

  const int w    = threadIdx.x >> 5;
  const int lane = threadIdx.x & 31;
  const int half = lane >> 4;        // which half-wave
  const int l    = lane & 15;
  const int row0 = (blockIdx.x * 8 + w) * 16;

  const float* xrow  = x  + (size_t)(row0 + l) * XCOLS;   // only first 1024 cols used
  const float* werow = We + (size_t)l * KIN;              // lane l -> N=l column of B

  // ---- embedding GEMM: K=1024, 2 WMMA k-steps per trip, loads front-loaded ----
  v8f c = {};
  for (int kb = 0; kb < KIN; kb += 64) {
    // ---- issue all global loads for both k-steps first (one big clause) ----
    // step 0 (K = kb .. kb+31)
    const float* ap0 = xrow + kb + half * 8;
    const float* ap1 = xrow + kb + 16 + half * 8;
    v4f xa0 = ldnt4(ap0), xa1 = ldnt4(ap0 + 4);
    v4f xa2 = ldnt4(ap1), xa3 = ldnt4(ap1 + 4);
    // step 1 (K = kb+32 .. kb+63)
    const float* ap2 = xrow + kb + 32 + half * 8;
    const float* ap3 = xrow + kb + 48 + half * 8;
    v4f xb0 = ldnt4(ap2), xb1 = ldnt4(ap2 + 4);
    v4f xb2 = ldnt4(ap3), xb3 = ldnt4(ap3 + 4);
    // We rows (L2-resident, regular temporal)
    const float* bp0 = werow + kb + half * 8;
    const float* bp1 = werow + kb + 32 + half * 8;
    v4f wa0 = ld4(bp0),      wa1 = ld4(bp0 + 4);
    v4f wa2 = ld4(bp0 + 16), wa3 = ld4(bp0 + 20);
    v4f wb0 = ld4(bp1),      wb1 = ld4(bp1 + 4);
    v4f wb2 = ld4(bp1 + 16), wb3 = ld4(bp1 + 20);

    // ---- step 0: cvt + WMMA ----
    V16U a, b;
    cvt8(a.e + 0, xa0, xa1);
    cvt8(a.e + 8, xa2, xa3);
    cvt8(b.e + 0, wa0, wa1);
    cvt8(b.e + 8, wa2, wa3);
    c = wmma_bf16(a.v, b.v, c);

    // ---- step 1: cvt + WMMA ----
    V16U a2, b2;
    cvt8(a2.e + 0, xb0, xb1);
    cvt8(a2.e + 8, xb2, xb3);
    cvt8(b2.e + 0, wb0, wb1);
    cvt8(b2.e + 8, wb2, wb3);
    c = wmma_bf16(a2.v, b2.v, c);
  }

  // ---- bias + relu; C layout: vgpr i -> (M = i + 8*half, N = l) ----
  float bn = be[l];
  #pragma unroll
  for (int i = 0; i < 8; i++) {
    float v = c[i] + bn;
    eLds[w][i + 8 * half][l] = v > 0.f ? v : 0.f;
  }
  __syncthreads();

  // ---- per-row work: lane pair (l, l+16) handles row m=l; experts 2*half,2*half+1
  float er[16];
  {
    const v4f* ep = (const v4f*)&eLds[w][l][0];
    v4f e0 = ep[0], e1 = ep[1], e2 = ep[2], e3 = ep[3];
    er[0]=e0[0]; er[1]=e0[1]; er[2]=e0[2]; er[3]=e0[3];
    er[4]=e1[0]; er[5]=e1[1]; er[6]=e1[2]; er[7]=e1[3];
    er[8]=e2[0]; er[9]=e2[1]; er[10]=e2[2]; er[11]=e2[3];
    er[12]=e3[0]; er[13]=e3[1]; er[14]=e3[2]; er[15]=e3[3];
  }

  #pragma unroll
  for (int g = 0; g < 2; g++) {
    int ge = 2 * half + g;
    float s = bg[ge];
    #pragma unroll
    for (int n = 0; n < 16; n++) s = fmaf(er[n], Wg[ge * 16 + n], s);
    gLds[w][l][ge] = s;
  }
  __syncthreads();

  float l0 = gLds[w][l][0], l1 = gLds[w][l][1];
  float l2 = gLds[w][l][2], l3 = gLds[w][l][3];
  float mx = fmaxf(fmaxf(l0, l1), fmaxf(l2, l3));
  float p0 = __expf(l0 - mx), p1 = __expf(l1 - mx);
  float p2 = __expf(l2 - mx), p3 = __expf(l3 - mx);
  float inv = 1.f / (p0 + p1 + p2 + p3);
  float gate[4] = {p0 * inv, p1 * inv, p2 * inv, p3 * inv};

  // ---- fc1 for two experts, scaled by gate, packed to bf16 ----
  union { __bf16 h[32]; uint4 q[4]; } hb;
  #pragma unroll
  for (int g = 0; g < 2; g++) {
    int ge = 2 * half + g;
    float gv = gate[ge];
    const float* w1p = W1 + (size_t)ge * (EMBED * TINY);
    float acc[16];
    #pragma unroll
    for (int t = 0; t < 16; t++) acc[t] = b1[ge * 16 + t];
    #pragma unroll
    for (int n = 0; n < 16; n++) {
      float en = er[n];
      union { v4f q[4]; float f[16]; } wr;
      const v4f* wp = (const v4f*)(w1p + n * 16);
      wr.q[0] = wp[0]; wr.q[1] = wp[1]; wr.q[2] = wp[2]; wr.q[3] = wp[3];
      #pragma unroll
      for (int t = 0; t < 16; t++) acc[t] = fmaf(en, wr.f[t], acc[t]);
    }
    #pragma unroll
    for (int t = 0; t < 16; t++) {
      float hv = acc[t] > 0.f ? acc[t] : 0.f;
      hb.h[g * 16 + t] = (__bf16)(gv * hv);
    }
  }

  // ---- emit hg row (row m=l): half0 -> cols 0..31, half1 -> cols 32..63 ----
  __bf16* hrow = hg + (size_t)(row0 + l) * KP;
  #pragma unroll
  for (int ch = 0; ch < 4; ch++)
    *(uint4*)(hrow + 32 * half + ch * 8) = hb.q[ch];
  // gate cols 64..67
  union { __bf16 e[2]; unsigned int u; } gp;
  gp.e[0] = (__bf16)gate[2 * half];
  gp.e[1] = (__bf16)gate[2 * half + 1];
  *(unsigned int*)(hrow + 64 + 2 * half) = gp.u;
  // zero padding cols 68..95
  uint4 z4 = {0, 0, 0, 0};
  if (half == 0) {
    uint2 z2 = {0, 0};
    *(uint2*)(hrow + 68) = z2;      // cols 68..71
    *(uint4*)(hrow + 72) = z4;      // cols 72..79
  } else {
    *(uint4*)(hrow + 80) = z4;      // cols 80..87
    *(uint4*)(hrow + 88) = z4;      // cols 88..95
  }
}

// --------------------------------------------------------------------------
// k2: out[B,1000] = hg[B,96] @ W2augT^T ; 3 WMMA k-steps per 16x16 tile.
// Block = one M-tile (16 rows); 8 waves sweep the 63 N-tiles.
// Output stores are non-temporal (written once, never re-read).
// --------------------------------------------------------------------------
__global__ __launch_bounds__(256) void k2_moe_gemm(
    const unsigned short* __restrict__ hg_raw,
    const unsigned short* __restrict__ w2t_raw,
    float* __restrict__ out)
{
  const __bf16* hg  = reinterpret_cast<const __bf16*>(hg_raw);
  const __bf16* w2t = reinterpret_cast<const __bf16*>(w2t_raw);

  const int w    = threadIdx.x >> 5;
  const int lane = threadIdx.x & 31;
  const int half = lane >> 4;
  const int l    = lane & 15;
  const int row0 = blockIdx.x * 16;

  // A fragments held in registers across the whole N sweep
  const __bf16* arow = hg + (size_t)(row0 + l) * KP;
  V16U a0, a1, a2;
  a0.q[0] = *(const uint4*)(arow + 0  + half * 8);
  a0.q[1] = *(const uint4*)(arow + 16 + half * 8);
  a1.q[0] = *(const uint4*)(arow + 32 + half * 8);
  a1.q[1] = *(const uint4*)(arow + 48 + half * 8);
  a2.q[0] = *(const uint4*)(arow + 64 + half * 8);
  a2.q[1] = *(const uint4*)(arow + 80 + half * 8);

  for (int nt = w; nt < NTILES; nt += 8) {
    int n0 = nt * 16;
    const __bf16* brow = w2t + (size_t)(n0 + l) * KP + half * 8;
    V16U b0, b1f, b2f;
    b0.q[0]  = *(const uint4*)(brow + 0);
    b0.q[1]  = *(const uint4*)(brow + 16);
    b1f.q[0] = *(const uint4*)(brow + 32);
    b1f.q[1] = *(const uint4*)(brow + 48);
    b2f.q[0] = *(const uint4*)(brow + 64);
    b2f.q[1] = *(const uint4*)(brow + 80);

    v8f c = {};
    c = wmma_bf16(a0.v, b0.v,  c);
    c = wmma_bf16(a1.v, b1f.v, c);
    c = wmma_bf16(a2.v, b2f.v, c);

    int col = n0 + l;
    if (col < OUTD) {
      float* op = out + (size_t)(row0 + 8 * half) * OUTD + col;
      #pragma unroll
      for (int i = 0; i < 8; i++)
        __builtin_nontemporal_store(c[i], op + (size_t)i * OUTD);
    }
  }
}

// --------------------------------------------------------------------------
extern "C" void kernel_launch(void* const* d_in, const int* in_sizes, int n_in,
                              void* d_out, int out_size, void* d_ws, size_t ws_size,
                              hipStream_t stream) {
  const float* x  = (const float*)d_in[0];
  const float* We = (const float*)d_in[1];
  const float* be = (const float*)d_in[2];
  const float* W1 = (const float*)d_in[3];
  const float* b1 = (const float*)d_in[4];
  const float* W2 = (const float*)d_in[5];
  const float* b2 = (const float*)d_in[6];
  const float* Wg = (const float*)d_in[7];
  const float* bg = (const float*)d_in[8];
  float* out = (float*)d_out;

  unsigned short* hg  = (unsigned short*)d_ws;                    // 32768*96 bf16 = 6.0 MiB
  unsigned short* w2t = (unsigned short*)((char*)d_ws + (size_t)BATCH * KP * 2);

  k0_pack_w2<<<(NTILES * 16 * KP + 255) / 256, 256, 0, stream>>>(W2, b2, w2t);
  k1_embed_gate_hg<<<BATCH / 16 / 8, 256, 0, stream>>>(x, We, be, W1, b1, Wg, bg, hg);
  k2_moe_gemm<<<BATCH / 16, 256, 0, stream>>>(hg, w2t, out);
}
